// AGCB_Element_11613591568418
// MI455X (gfx1250) — matrix operand order repaired and checked by reading, not compiled, not run on hardware
//
#include <hip/hip_runtime.h>
#include <math.h>

// ---- problem constants ----
#define BATCH 2
#define CH    64
#define HH    128
#define WW    128
#define HB    64            // block height (s=2)
#define NPIX  4096          // pixels per quadrant block
#define NBB   8             // 4 quadrants * 2 batch
#define NCHUNK 128          // 4096 / 32 key chunks

typedef __attribute__((ext_vector_type(16))) _Float16 v16h;
typedef __attribute__((ext_vector_type(16))) __bf16   v16bf;
typedef __attribute__((ext_vector_type(8)))  float    v8f;

// ---- workspace layout (bytes) ----
#define OFF_POOL  0u          // 2*64*4 f32            = 2048
#define OFF_GCA   2048u       // 2*64*4 f32            = 2048
#define OFF_Q     4096u       // 8*2*4096 f32          = 262144
#define OFF_K     266240u     // 8*2*4096 f32          = 262144
#define OFF_V     528384u     // 8*128*4*32*16 f16     = 4194304
#define OFF_WA    4722688u    // 9*4*2*32*16 bf16      = 73728
#define OFF_CCTX  4759552u    // 2*128*128*64 bf16     = 4194304
// total ~8.95 MB

// ------------------------------------------------------------------
// 1) adaptive 2x2 max pool: one WG per output element
__global__ void pool_kernel(const float* __restrict__ x, float* __restrict__ pooled) {
  __shared__ float red[256];
  int idx = blockIdx.x;              // b*256 + c*4 + sy*2 + sx
  int b = idx >> 8, rem = idx & 255;
  int c = rem >> 2, sy = (rem >> 1) & 1, sx = rem & 1;
  float m = -1e30f;
  for (int e = threadIdx.x; e < 4096; e += 256) {
    int h = sy * HB + (e >> 6);
    int w = sx * HB + (e & 63);
    m = fmaxf(m, x[(((size_t)b * CH + c) * HH + h) * WW + w]);
  }
  red[threadIdx.x] = m;
  __syncthreads();
  for (int s = 128; s > 0; s >>= 1) {
    if (threadIdx.x < s) red[threadIdx.x] = fmaxf(red[threadIdx.x], red[threadIdx.x + s]);
    __syncthreads();
  }
  if (threadIdx.x == 0) pooled[idx] = red[0];
}

// ------------------------------------------------------------------
// 2) tiny non-local on the pooled 2x2 (N=4), one thread per (b,c)
__global__ void gca_nl_kernel(const float* __restrict__ pooled,
                              const float* __restrict__ qw, const float* __restrict__ qb,
                              const float* __restrict__ kw, const float* __restrict__ kb,
                              const float* __restrict__ vw, const float* __restrict__ vb,
                              const float* __restrict__ gamma,
                              float* __restrict__ gca_small) {
  int t = threadIdx.x;
  if (t >= 128) return;
  int b = t >> 6, c = t & 63;
  const float* P = pooled + b * 256;           // [cc][4]
  float q[2][4], k[2][4];
  for (int i = 0; i < 2; ++i)
    for (int n = 0; n < 4; ++n) {
      float aq = qb[i], ak = kb[i];
      for (int cc = 0; cc < 64; ++cc) {
        float pv = P[cc * 4 + n];
        aq = fmaf(qw[i * 64 + cc], pv, aq);
        ak = fmaf(kw[i * 64 + cc], pv, ak);
      }
      q[i][n] = aq; k[i][n] = ak;
    }
  float vv[4];
  for (int n = 0; n < 4; ++n) {
    float a = vb[c];
    for (int cc = 0; cc < 64; ++cc) a = fmaf(vw[c * 64 + cc], P[cc * 4 + n], a);
    vv[n] = a;
  }
  float gg = gamma[0];
  for (int n = 0; n < 4; ++n) {
    float s[4], mx = -1e30f;
    for (int m = 0; m < 4; ++m) { s[m] = q[0][n] * k[0][m] + q[1][n] * k[1][m]; mx = fmaxf(mx, s[m]); }
    float den = 0.f;
    for (int m = 0; m < 4; ++m) { s[m] = __expf(s[m] - mx); den += s[m]; }
    float o = 0.f;
    for (int m = 0; m < 4; ++m) o += vv[m] * s[m];
    o /= den;
    gca_small[b * 256 + c * 4 + n] = gg * o + P[c * 4 + n];
  }
}

// ------------------------------------------------------------------
// 3) q,k 1x1 conv (2 channels each) per block-batch
__global__ void qk_kernel(const float* __restrict__ x,
                          const float* __restrict__ qw, const float* __restrict__ qb,
                          const float* __restrict__ kw, const float* __restrict__ kb,
                          float* __restrict__ qws, float* __restrict__ kws) {
  int g = blockIdx.x * 256 + threadIdx.x;     // 8*4096
  int bb = g >> 12, n = g & 4095;
  int b = bb & 1, blk = bb >> 1, sy = blk >> 1, sx = blk & 1;
  int h = sy * HB + (n >> 6), w = sx * HB + (n & 63);
  float q0 = qb[0], q1 = qb[1], k0 = kb[0], k1 = kb[1];
  for (int cc = 0; cc < 64; ++cc) {
    float xv = x[(((size_t)b * CH + cc) * HH + h) * WW + w];
    q0 = fmaf(qw[cc], xv, q0);      q1 = fmaf(qw[64 + cc], xv, q1);
    k0 = fmaf(kw[cc], xv, k0);      k1 = fmaf(kw[64 + cc], xv, k1);
  }
  qws[bb * 8192 + n] = q0; qws[bb * 8192 + 4096 + n] = q1;
  kws[bb * 8192 + n] = k0; kws[bb * 8192 + 4096 + n] = k1;
}

// ------------------------------------------------------------------
// 4) v 1x1 conv -> f16 directly in WMMA B-matrix (32x16) per-lane layout
//    layout: [bb][chunk][ctile][lane][j], elem = v(c=ctile*16+lane%16,
//             m = chunk*32 + 16*(lane/16) + j)
__global__ void v_kernel(const float* __restrict__ x,
                         const float* __restrict__ vw, const float* __restrict__ vb,
                         _Float16* __restrict__ vB) {
  int g = blockIdx.x * 256 + threadIdx.x;     // 2,097,152
  int j = g & 15, lane = (g >> 4) & 31, ct = (g >> 9) & 3;
  int chunk = (g >> 11) & 127, bb = g >> 18;
  int c = ct * 16 + (lane & 15);
  int m = chunk * 32 + 16 * (lane >> 4) + j;
  int b = bb & 1, blk = bb >> 1, sy = blk >> 1, sx = blk & 1;
  int h = sy * HB + (m >> 6), w = sx * HB + (m & 63);
  float a = vb[c];
  for (int cc = 0; cc < 64; ++cc)
    a = fmaf(vw[c * 64 + cc], x[(((size_t)b * CH + cc) * HH + h) * WW + w], a);
  vB[g] = (_Float16)a;
}

// ------------------------------------------------------------------
// 5) conv weights -> bf16 in WMMA A-matrix (16x32) per-lane layout
//    layout: [tap][mtile][kchunk][lane][j]
__global__ void wa_kernel(const float* __restrict__ cw, __bf16* __restrict__ wA) {
  int g = blockIdx.x * 256 + threadIdx.x;     // 36864
  int j = g & 15, lane = (g >> 4) & 31, kc = (g >> 9) & 1;
  int mt = (g >> 10) & 3, tap = g >> 12;
  int half = lane >> 4;
  int o = mt * 16 + (lane & 15);
  int kk = 16 * half + j;
  int c = kc * 32 + kk;
  int ky = tap / 3, kx = tap % 3;
  wA[g] = (__bf16)cw[((o * 64 + c) * 3 + ky) * 3 + kx];
}

// ------------------------------------------------------------------
// 6) attention: per wave 16 query rows, two-pass softmax (head dim 2),
//    P.V^T via v_wmma_f32_16x16x32_f16; epilogue applies nl_gamma, residual,
//    bilinear+sigmoid GCA gate, writes NHWC bf16 conv input.
//    Softmax stats passes use float4 key blocks: lane-half h owns
//    m in [8i+4h, 8i+4h+4) -> 2x global_load_b128 per 4 keys.
__global__ void attn_kernel(const float* __restrict__ x,
                            const float* __restrict__ qws, const float* __restrict__ kws,
                            const _Float16* __restrict__ vB,
                            const float* __restrict__ nl_gamma,
                            const float* __restrict__ gca_small,
                            __bf16* __restrict__ cctx) {
  int ntile = blockIdx.x;                      // 0..31
  int bb    = blockIdx.y;                      // 0..7
  int wave = threadIdx.x >> 5, lane = threadIdx.x & 31;
  int half = lane >> 4;
  int nrow = ntile * 128 + wave * 16 + (lane & 15);   // row this lane owns (A layout)

  const float* qp = qws + bb * 8192;
  const float* k0p = kws + bb * 8192;
  const float* k1p = k0p + 4096;
  float q0 = qp[nrow], q1 = qp[4096 + nrow];

  // pass 1a: row max over all keys (float4 blocks, halves interleave blocks)
  const float4* k0v = (const float4*)k0p;
  const float4* k1v = (const float4*)k1p;
  float mx = -1e30f;
  for (int i = half; i < 1024; i += 2) {
    float4 a0 = k0v[i], a1 = k1v[i];
    mx = fmaxf(mx, fmaf(q0, a0.x, q1 * a1.x));
    mx = fmaxf(mx, fmaf(q0, a0.y, q1 * a1.y));
    mx = fmaxf(mx, fmaf(q0, a0.z, q1 * a1.z));
    mx = fmaxf(mx, fmaf(q0, a0.w, q1 * a1.w));
  }
  mx = fmaxf(mx, __shfl_xor(mx, 16, 32));

  // pass 1b: row sum of exp
  float sum = 0.f;
  for (int i = half; i < 1024; i += 2) {
    float4 a0 = k0v[i], a1 = k1v[i];
    sum += __expf(fmaf(q0, a0.x, q1 * a1.x) - mx);
    sum += __expf(fmaf(q0, a0.y, q1 * a1.y) - mx);
    sum += __expf(fmaf(q0, a0.z, q1 * a1.z) - mx);
    sum += __expf(fmaf(q0, a0.w, q1 * a1.w) - mx);
  }
  sum += __shfl_xor(sum, 16, 32);
  float inv = 1.0f / sum;

  // pass 2: P tile (A layout) x V^T (pre-laid-out B tiles) -> f32 accum
  v8f acc[4];
  #pragma unroll
  for (int ct = 0; ct < 4; ++ct)
    #pragma unroll
    for (int r = 0; r < 8; ++r) acc[ct][r] = 0.f;

  const v16h* vb16 = (const v16h*)vB;
  for (int chunk = 0; chunk < NCHUNK; ++chunk) {
    v16h a;
    #pragma unroll
    for (int j = 0; j < 16; ++j) {
      int m = chunk * 32 + 16 * half + j;
      float s = fmaf(q0, k0p[m], q1 * k1p[m]);
      a[j] = (_Float16)(__expf(s - mx) * inv);
    }
    size_t base = ((size_t)(bb * 128 + chunk)) * 4 * 32;
    #pragma unroll
    for (int ct = 0; ct < 4; ++ct) {
      v16h bmat = vb16[base + (size_t)ct * 32 + lane];
      acc[ct] = __builtin_amdgcn_wmma_f32_16x16x32_f16(false, a, false, bmat,
                                                       (short)0, acc[ct], false, false);
    }
  }

  // epilogue: D layout M = r + 8*half (query row), N = lane%16 (channel)
  int b = bb & 1, blk = bb >> 1, sy = blk >> 1, sx = blk & 1;
  float gnl = nl_gamma[0];
  #pragma unroll
  for (int ct = 0; ct < 4; ++ct) {
    int c = ct * 16 + (lane & 15);
    const float* gp = gca_small + (b * 64 + c) * 4;   // 2x2 gate values
    float g00 = gp[0], g01 = gp[1], g10 = gp[2], g11 = gp[3];
    #pragma unroll
    for (int r = 0; r < 8; ++r) {
      int nd = ntile * 128 + wave * 16 + r + 8 * half;
      int h = sy * HB + (nd >> 6);
      int w = sx * HB + (nd & 63);
      float fy = h * (1.0f / 127.0f); int y0 = (int)fy; if (y0 > 1) y0 = 1; float wy = fy - y0;
      float fx = w * (1.0f / 127.0f); int x0 = (int)fx; if (x0 > 1) x0 = 1; float wx = fx - x0;
      int y1 = y0 < 1 ? y0 + 1 : 1, x1 = x0 < 1 ? x0 + 1 : 1;
      float gA = (y0 == 0) ? g00 : g10, gB = (y0 == 0) ? g01 : g11;
      float gC = (y1 == 0) ? g00 : g10, gD = (y1 == 0) ? g01 : g11;
      float top = (x0 == 0 ? gA : gB) * (1.f - wx) + (x1 == 0 ? gA : gB) * wx;
      float bot = (x0 == 0 ? gC : gD) * (1.f - wx) + (x1 == 0 ? gC : gD) * wx;
      float gate = 1.0f / (1.0f + __expf(-(top * (1.f - wy) + bot * wy)));
      float xin = x[(((size_t)b * CH + c) * HH + h) * WW + w];
      float ctxv = gnl * acc[ct][r] + xin;
      cctx[(((size_t)b * HH + h) * WW + w) * 64 + c] = (__bf16)(ctxv * gate);
    }
  }
}

// ------------------------------------------------------------------
// 7) 3x3 conv as implicit GEMM (v_wmma_f32_16x16x32_bf16) + BN + residual + relu
//    one wave per 16-pixel row strip; 4 M-tiles (64 out ch), 2 K-chunks, 9 taps
__global__ void conv_kernel(const __bf16* __restrict__ cctx, const __bf16* __restrict__ wA,
                            const float* __restrict__ cb,
                            const float* __restrict__ bnw, const float* __restrict__ bnb,
                            const float* __restrict__ bnm, const float* __restrict__ bnv,
                            const float* __restrict__ gamma,
                            const float* __restrict__ x, float* __restrict__ out) {
  int wave = threadIdx.x >> 5, lane = threadIdx.x & 31;
  int half = lane >> 4;
  int sid = blockIdx.x * 8 + wave;             // 2048 strips
  int b = sid >> 10, rem = sid & 1023;
  int h = rem >> 3, w0 = (rem & 7) << 4;

  v8f acc[4];
  #pragma unroll
  for (int mt = 0; mt < 4; ++mt)
    #pragma unroll
    for (int r = 0; r < 8; ++r) acc[mt][r] = 0.f;

  const v16bf* wAp = (const v16bf*)wA;
  #pragma unroll
  for (int tap = 0; tap < 9; ++tap) {
    int ky = tap / 3 - 1, kx = tap % 3 - 1;
    int hh = h + ky;
    int ww = w0 + (lane & 15) + kx;
    bool inb = (hh >= 0) & (hh < HH) & (ww >= 0) & (ww < WW);
    v16bf btile[2];
    #pragma unroll
    for (int kc = 0; kc < 2; ++kc) {
      v16bf bv;
      #pragma unroll
      for (int j = 0; j < 16; ++j) bv[j] = (__bf16)0.0f;
      if (inb) {
        int cbase = kc * 32 + 16 * half;
        bv = *(const v16bf*)(cctx + (((size_t)b * HH + hh) * WW + ww) * 64 + cbase);
      }
      btile[kc] = bv;
    }
    #pragma unroll
    for (int mt = 0; mt < 4; ++mt)
      #pragma unroll
      for (int kc = 0; kc < 2; ++kc) {
        v16bf aA = wAp[(size_t)((tap * 4 + mt) * 2 + kc) * 32 + lane];
        acc[mt] = __builtin_amdgcn_wmma_f32_16x16x32_bf16(false, aA, false, btile[kc],
                                                          (short)0, acc[mt], false, false);
      }
  }

  float gam = gamma[0];
  int wpix = w0 + (lane & 15);
  #pragma unroll
  for (int mt = 0; mt < 4; ++mt)
    #pragma unroll
    for (int r = 0; r < 8; ++r) {
      int o = mt * 16 + r + 8 * half;
      float val = acc[mt][r] + cb[o];
      val = (val - bnm[o]) * rsqrtf(bnv[o] + 1e-5f) * bnw[o] + bnb[o];
      size_t oi = (((size_t)b * CH + o) * HH + h) * WW + wpix;
      out[oi] = fmaxf(gam * val + x[oi], 0.f);
    }
}

// ------------------------------------------------------------------
extern "C" void kernel_launch(void* const* d_in, const int* in_sizes, int n_in,
                              void* d_out, int out_size, void* d_ws, size_t ws_size,
                              hipStream_t stream) {
  const float* x       = (const float*)d_in[0];
  const float* nl_q_w  = (const float*)d_in[1];
  const float* nl_q_b  = (const float*)d_in[2];
  const float* nl_k_w  = (const float*)d_in[3];
  const float* nl_k_b  = (const float*)d_in[4];
  const float* nl_v_w  = (const float*)d_in[5];
  const float* nl_v_b  = (const float*)d_in[6];
  const float* nl_g    = (const float*)d_in[7];
  const float* gca_q_w = (const float*)d_in[8];
  const float* gca_q_b = (const float*)d_in[9];
  const float* gca_k_w = (const float*)d_in[10];
  const float* gca_k_b = (const float*)d_in[11];
  const float* gca_v_w = (const float*)d_in[12];
  const float* gca_v_b = (const float*)d_in[13];
  const float* gca_g   = (const float*)d_in[14];
  const float* conv_w  = (const float*)d_in[15];
  const float* conv_b  = (const float*)d_in[16];
  const float* bn_w    = (const float*)d_in[17];
  const float* bn_b    = (const float*)d_in[18];
  const float* bn_mean = (const float*)d_in[19];
  const float* bn_var  = (const float*)d_in[20];
  const float* gamma   = (const float*)d_in[21];

  char* ws = (char*)d_ws;
  float*    pooled    = (float*)(ws + OFF_POOL);
  float*    gca_small = (float*)(ws + OFF_GCA);
  float*    qws       = (float*)(ws + OFF_Q);
  float*    kws       = (float*)(ws + OFF_K);
  _Float16* vB        = (_Float16*)(ws + OFF_V);
  __bf16*   wA        = (__bf16*)(ws + OFF_WA);
  __bf16*   cctx      = (__bf16*)(ws + OFF_CCTX);

  pool_kernel<<<512, 256, 0, stream>>>(x, pooled);
  gca_nl_kernel<<<1, 128, 0, stream>>>(pooled, gca_q_w, gca_q_b, gca_k_w, gca_k_b,
                                       gca_v_w, gca_v_b, gca_g, gca_small);
  qk_kernel<<<128, 256, 0, stream>>>(x, nl_q_w, nl_q_b, nl_k_w, nl_k_b, qws, kws);
  v_kernel<<<8192, 256, 0, stream>>>(x, nl_v_w, nl_v_b, vB);
  wa_kernel<<<144, 256, 0, stream>>>(conv_w, wA);
  attn_kernel<<<dim3(32, 8), 256, 0, stream>>>(x, qws, kws, vB, nl_g, gca_small, cctx);
  conv_kernel<<<256, 256, 0, stream>>>(cctx, wA, conv_b, bn_w, bn_b, bn_mean, bn_var,
                                       gamma, x, d_out ? (float*)d_out : nullptr);
}
